// MoELayer_43662637532108
// MI455X (gfx1250) — matrix operand order repaired and checked by reading, not compiled, not run on hardware
//
#include <hip/hip_runtime.h>
#include <hip/hip_bf16.h>

// Problem constants (from reference): B=2,S=2048 -> T=4096 tokens
#define TOK 4096
#define DM  1024
#define FM  4096
#define EM  8
#define KM  2

typedef __attribute__((ext_vector_type(16))) __bf16 v16bf;
typedef __attribute__((ext_vector_type(8)))  float  v8f;

union FragBF { uint4 q[2]; v16bf v; };

// A-fragment (16x32 bf16, wave32): lane l -> row M=(l&15), g=(l>>4).
// elements 0..7 = K in [k0+g*8, +8), elements 8..15 = K in [k0+16+g*8, +8)
static __device__ __forceinline__ v16bf ld_frag_a(const __bf16* row, int k0, int g) {
    FragBF f;
    f.q[0] = *(const uint4*)(row + k0 + g * 8);
    f.q[1] = *(const uint4*)(row + k0 + 16 + g * 8);
    return f.v;
}

// B-fragment (32x16 bf16, wave32): lane l -> col N=(l&15), g=(l>>4).
// elements 0..7 = K in [k0+g*8, +8), elements 8..15 = K in [k0+16+g*8, +8). `row` points at column-major
// (N-major) LDS staging: row = &wl[n * stride].
static __device__ __forceinline__ v16bf ld_frag_b(const __bf16* row, int k0, int g) {
    FragBF f;
    f.q[0] = *(const uint4*)(row + k0 + g * 8);
    f.q[1] = *(const uint4*)(row + k0 + 16 + g * 8);
    return f.v;
}

static __device__ __forceinline__ v8f wmma_bf16(v16bf a, v16bf b, v8f c) {
    return __builtin_amdgcn_wmma_f32_16x16x32_bf16(false, a, false, b, (short)0, c, false, false);
}

// ---------------------------------------------------------------- zero init
__global__ __launch_bounds__(256) void k_zero(float* __restrict__ out, int* __restrict__ counts) {
    int i = blockIdx.x * 256 + threadIdx.x;
    if (i < TOK * DM) out[i] = 0.0f;
    if (blockIdx.x == 0 && threadIdx.x < 2 * EM) counts[threadIdx.x] = 0; // counts[8] + pos[8]
}

// ---------------------------------------------------------------- router
// one wave (32 lanes) per token
__global__ __launch_bounds__(256) void k_router(const float* __restrict__ x,
                                                const float* __restrict__ Wr,
                                                const float* __restrict__ br,
                                                float* __restrict__ probs_out,
                                                float* __restrict__ cw,
                                                int* __restrict__ topi,
                                                int* __restrict__ counts) {
    const int wave = threadIdx.x >> 5;
    const int lane = threadIdx.x & 31;
    const int t = blockIdx.x * 8 + wave;
    if (t >= TOK) return;

    float acc[EM];
#pragma unroll
    for (int e = 0; e < EM; e++) acc[e] = 0.0f;

    const float* xr = x + (size_t)t * DM;
    for (int c = lane; c < DM; c += 32) {
        float xv = xr[c];
        const float4* wr4 = (const float4*)(Wr + (size_t)c * EM);
        float4 a = wr4[0];
        float4 b = wr4[1];
        acc[0] += xv * a.x; acc[1] += xv * a.y; acc[2] += xv * a.z; acc[3] += xv * a.w;
        acc[4] += xv * b.x; acc[5] += xv * b.y; acc[6] += xv * b.z; acc[7] += xv * b.w;
    }
#pragma unroll
    for (int e = 0; e < EM; e++) {
        float v = acc[e];
#pragma unroll
        for (int s = 16; s > 0; s >>= 1) v += __shfl_xor(v, s, 32);
        acc[e] = v;
    }
    if (lane == 0) {
        float lg[EM];
        float mx = -3.4e38f;
#pragma unroll
        for (int e = 0; e < EM; e++) { lg[e] = acc[e] + br[e]; mx = fmaxf(mx, lg[e]); }
        float sum = 0.0f;
        float p[EM];
#pragma unroll
        for (int e = 0; e < EM; e++) { p[e] = expf(lg[e] - mx); sum += p[e]; }
        float inv = 1.0f / sum;
#pragma unroll
        for (int e = 0; e < EM; e++) {
            p[e] *= inv;
            probs_out[(size_t)t * EM + e] = p[e];
        }
        // top-2 (first-occurrence ties like jax top_k)
        int i1 = 0; float p1 = p[0];
#pragma unroll
        for (int e = 1; e < EM; e++) if (p[e] > p1) { p1 = p[e]; i1 = e; }
        int i2 = -1; float p2 = -1.0f;
#pragma unroll
        for (int e = 0; e < EM; e++) if (e != i1 && p[e] > p2) { p2 = p[e]; i2 = e; }
#pragma unroll
        for (int e = 0; e < EM; e++) {
            float w = (e == i1) ? p1 : ((e == i2) ? p2 : 0.0f);
            cw[(size_t)t * EM + e] = w;
        }
        topi[t * 2 + 0] = i1;
        topi[t * 2 + 1] = i2;
        atomicAdd(&counts[i1], 1);
        atomicAdd(&counts[i2], 1);
    }
}

// ---------------------------------------------------------------- tiny scan
__global__ void k_scan(const int* __restrict__ counts, int* __restrict__ offs, int* __restrict__ pos) {
    if (threadIdx.x == 0 && blockIdx.x == 0) {
        int o = 0;
        for (int e = 0; e < EM; e++) { offs[e] = o; pos[e] = o; o += counts[e]; }
        offs[EM] = o;
    }
}

// ---------------------------------------------------------------- gather assignments
__global__ __launch_bounds__(256) void k_gather(const int* __restrict__ topi,
                                                int* __restrict__ pos,
                                                int* __restrict__ tok) {
    int t = blockIdx.x * 256 + threadIdx.x;
    if (t >= TOK) return;
#pragma unroll
    for (int k = 0; k < KM; k++) {
        int e = topi[t * 2 + k];
        int slot = atomicAdd(&pos[e], 1);
        tok[slot] = t;
    }
}

// ---------------------------------------------------------------- fused expert FFN
// grid: 8 experts * 64 token-tiles of 64; block: 256 threads (8 waves)
// LDS (dynamic):
//   xs : [64][1032]  bf16  gathered X tile              132096 B
//   wl : B staging; GEMM1 layout [64 f][1032 k] bf16,   132096 B
//        GEMM2 layout [1024 d][40 k] bf16 (reuses wl)
//   hs : [64][72]    bf16  h chunk                        9216 B
//   ts : [64] int, cv : [64] float                         512 B
#define XS_STR 1032
#define WL_STR1 1032
#define WL_STR2 40
#define HS_STR 72
#define SMEM_BYTES (132096 + 132096 + 9216 + 256 + 256)

__global__ __launch_bounds__(256, 1) void k_ffn(const float* __restrict__ x,
                                                const float* __restrict__ W1,
                                                const float* __restrict__ b1,
                                                const float* __restrict__ W2,
                                                const float* __restrict__ b2,
                                                const float* __restrict__ cw,
                                                const int* __restrict__ counts,
                                                const int* __restrict__ offs,
                                                const int* __restrict__ tok,
                                                float* __restrict__ out) {
    extern __shared__ char smem[];
    __bf16* xs = (__bf16*)smem;
    __bf16* wl = (__bf16*)(smem + 132096);
    __bf16* hs = (__bf16*)(smem + 132096 + 132096);
    int*    ts = (int*)  (smem + 132096 + 132096 + 9216);
    float*  cv = (float*)(smem + 132096 + 132096 + 9216 + 256);

    const int e = blockIdx.x >> 6;
    const int mt = blockIdx.x & 63;
    const int cnt = counts[e];
    const int m0 = mt * 64;
    if (m0 >= cnt) return;
    const int off = offs[e];
    const int tid = threadIdx.x;

    if (tid < 64) {
        int rg = m0 + tid;
        int tk = (rg < cnt) ? tok[off + rg] : -1;
        ts[tid] = tk;
        cv[tid] = (tk >= 0) ? cw[(size_t)tk * EM + e] : 0.0f;
    }
    __syncthreads();

    // gather X rows -> bf16 LDS
    for (int idx = tid; idx < 64 * DM; idx += 256) {
        int r = idx >> 10, c = idx & (DM - 1);
        int tk = ts[r];
        float v = (tk >= 0) ? x[(size_t)tk * DM + c] : 0.0f;
        xs[r * XS_STR + c] = (__bf16)v;
    }

    const int wv = tid >> 5, ln = tid & 31;
    const int g = ln >> 4, lm = ln & 15;

    // GEMM2 accumulators: this wave owns output cols [wv*128, wv*128+128)
    v8f acc[4][8];
#pragma unroll
    for (int i = 0; i < 4; i++)
#pragma unroll
        for (int j = 0; j < 8; j++) acc[i][j] = (v8f){0,0,0,0,0,0,0,0};

    const float* W1e = W1 + (size_t)e * DM * FM;
    const float* W2e = W2 + (size_t)e * FM * DM;

    // GEMM1 wave tile: mi = wv&3 (16 rows), nbase = (wv>>2)*2 (two 16-col tiles)
    const int mi = wv & 3;
    const int nbase = (wv >> 2) * 2;

    for (int fc = 0; fc < FM; fc += 64) {
        __syncthreads(); // wl free; xs ready (first iter)
        // stage W1[:, fc..fc+64) -> wl[f][k] (f-major for B fragments)
        for (int idx = tid; idx < 64 * DM; idx += 256) {
            int k = idx >> 6, f = idx & 63;
            wl[f * WL_STR1 + k] = (__bf16)(W1e[(size_t)k * FM + fc + f]);
        }
        __syncthreads();

        // ---- GEMM1: h[64,64] = relu(Xs[64,1024] @ W1chunk + b1)
        v8f h0 = (v8f){0,0,0,0,0,0,0,0};
        v8f h1 = (v8f){0,0,0,0,0,0,0,0};
        const __bf16* arow  = xs + (mi * 16 + lm) * XS_STR;
        const __bf16* brow0 = wl + (nbase * 16 + lm) * WL_STR1;
        const __bf16* brow1 = brow0 + 16 * WL_STR1;
#pragma unroll 4
        for (int kb = 0; kb < DM; kb += 32) {
            v16bf a  = ld_frag_a(arow, kb, g);
            v16bf bf0 = ld_frag_b(brow0, kb, g);
            v16bf bf1 = ld_frag_b(brow1, kb, g);
            h0 = wmma_bf16(a, bf0, h0);
            h1 = wmma_bf16(a, bf1, h1);
        }
        float bb0 = b1[(size_t)e * FM + fc + nbase * 16 + lm];
        float bb1 = b1[(size_t)e * FM + fc + (nbase + 1) * 16 + lm];
#pragma unroll
        for (int v = 0; v < 8; v++) {
            int r = mi * 16 + g * 8 + v;
            hs[r * HS_STR + nbase * 16 + lm]       = (__bf16)fmaxf(h0[v] + bb0, 0.0f);
            hs[r * HS_STR + (nbase + 1) * 16 + lm] = (__bf16)fmaxf(h1[v] + bb1, 0.0f);
        }
        __syncthreads(); // hs visible; wl free (GEMM1 done reading)

        // ---- GEMM2: y[64,1024] += h[64,64] @ W2[fc..fc+64, :]
#pragma unroll
        for (int ks = 0; ks < 2; ks++) {
            // stage W2[fc+ks*32 .. +32, :] -> wl[d][k] (d-major for B fragments)
            for (int idx = tid; idx < 32 * DM; idx += 256) {
                int kk = idx >> 10, d = idx & (DM - 1);
                wl[d * WL_STR2 + kk] = (__bf16)(W2e[(size_t)(fc + ks * 32 + kk) * DM + d]);
            }
            __syncthreads();
            v16bf af[4];
#pragma unroll
            for (int mq = 0; mq < 4; mq++)
                af[mq] = ld_frag_a(hs + (mq * 16 + lm) * HS_STR, ks * 32, g);
#pragma unroll
            for (int nj = 0; nj < 8; nj++) {
                v16bf bfr = ld_frag_b(wl + (wv * 128 + nj * 16 + lm) * WL_STR2, 0, g);
#pragma unroll
                for (int mq = 0; mq < 4; mq++)
                    acc[mq][nj] = wmma_bf16(af[mq], bfr, acc[mq][nj]);
            }
            __syncthreads(); // wl free for next stage
        }
    }

    // ---- weighted scatter-add: out[t,:] += cw * (y + b2[e,:])
#pragma unroll
    for (int nj = 0; nj < 8; nj++) {
        int d = wv * 128 + nj * 16 + lm;
        float b2v = b2[(size_t)e * DM + d];
#pragma unroll
        for (int mq = 0; mq < 4; mq++) {
#pragma unroll
            for (int v = 0; v < 8; v++) {
                int r = mq * 16 + g * 8 + v;
                int tk = ts[r];
                float w = cv[r];
                if (tk >= 0 && w != 0.0f)
                    atomicAdd(&out[(size_t)tk * DM + d], w * (acc[mq][nj][v] + b2v));
            }
        }
    }
}

// ---------------------------------------------------------------- launch
extern "C" void kernel_launch(void* const* d_in, const int* in_sizes, int n_in,
                              void* d_out, int out_size, void* d_ws, size_t ws_size,
                              hipStream_t stream) {
    (void)in_sizes; (void)n_in; (void)out_size; (void)ws_size;
    const float* x  = (const float*)d_in[0];
    const float* Wr = (const float*)d_in[1];
    const float* br = (const float*)d_in[2];
    const float* W1 = (const float*)d_in[3];
    const float* b1 = (const float*)d_in[4];
    const float* W2 = (const float*)d_in[5];
    const float* b2 = (const float*)d_in[6];

    float* out   = (float*)d_out;
    float* probs = out + (size_t)TOK * DM;

    // workspace layout
    float* cw    = (float*)d_ws;               // TOK*EM
    int*   topi  = (int*)(cw + TOK * EM);      // TOK*KM
    int*   counts = (int*)(topi + TOK * KM);   // EM
    int*   pos    = counts + EM;               // EM
    int*   offs   = pos + EM;                  // EM+1
    int*   tok    = offs + EM + 1;             // TOK*KM

    k_zero<<<(TOK * DM + 255) / 256, 256, 0, stream>>>(out, counts);
    k_router<<<TOK / 8, 256, 0, stream>>>(x, Wr, br, probs, cw, topi, counts);
    k_scan<<<1, 32, 0, stream>>>(counts, offs, pos);
    k_gather<<<TOK / 256, 256, 0, stream>>>(topi, pos, tok);
    k_ffn<<<EM * 64, 256, SMEM_BYTES, stream>>>(x, W1, b1, W2, b2, cw, counts, offs, tok, out);
}
